// AttentionLayer_72353019068602
// MI455X (gfx1250) — matrix operand order repaired and checked by reading, not compiled, not run on hardware
//
#include <hip/hip_runtime.h>
#include <math.h>

// ---------------------------------------------------------------------------
// Problem constants (match the reference)
// ---------------------------------------------------------------------------
constexpr int kB   = 32;
constexpr int kTE  = 2048;
constexpr int kTD  = 64;
constexpr int kD   = 1024;

constexpr int kKsteps = kD / 32;   // 32 K-steps of 32 for the f16 WMMA
constexpr int kNtiles = kD / 16;   // 64 column tiles of 16
constexpr int kMtiles = 4;         // M-tiles per workgroup
constexpr int kMrows  = kMtiles * 16;          // 64 rows (TE dim) per workgroup
constexpr int kWGs    = kB * (kTE / kMrows);   // 1024 workgroups in K2

typedef __attribute__((ext_vector_type(16))) _Float16 v16h;
typedef __attribute__((ext_vector_type(8)))  _Float16 v8h;
typedef __attribute__((ext_vector_type(8)))  float    v8f;

// ---------------------------------------------------------------------------
// K0: convert W_a (f32, row-major D x D) into f16 pre-tiled B-fragment order:
//     Wt[((kstep*64 + ntile)*32 + kk)*16 + nn] = W[kstep*32+kk][ntile*16+nn]
// so a wave's B fragment load is one contiguous 1 KB block (fully coalesced,
// L2-resident 2 MB). 65536 jobs: one job converts one 16-wide row slice.
// ---------------------------------------------------------------------------
__global__ __launch_bounds__(256)
void k0_tile_w(const float* __restrict__ W, _Float16* __restrict__ Wt) {
    int job   = blockIdx.x * 256 + threadIdx.x;   // 1024 * 64 jobs
    int k     = job >> 6;                          // 0..1023
    int ntile = job & 63;                          // 0..63
    const float4* src = (const float4*)(W + (size_t)k * kD + ntile * 16);
    float4 f0 = src[0], f1 = src[1], f2 = src[2], f3 = src[3];
    v8h lo, hi;
    lo[0]=(_Float16)f0.x; lo[1]=(_Float16)f0.y; lo[2]=(_Float16)f0.z; lo[3]=(_Float16)f0.w;
    lo[4]=(_Float16)f1.x; lo[5]=(_Float16)f1.y; lo[6]=(_Float16)f1.z; lo[7]=(_Float16)f1.w;
    hi[0]=(_Float16)f2.x; hi[1]=(_Float16)f2.y; hi[2]=(_Float16)f2.z; hi[3]=(_Float16)f2.w;
    hi[4]=(_Float16)f3.x; hi[5]=(_Float16)f3.y; hi[6]=(_Float16)f3.z; hi[7]=(_Float16)f3.w;
    int kstep = k >> 5, kk = k & 31;
    _Float16* dst = Wt + ((((size_t)kstep * kNtiles + ntile) * 32 + kk) << 4);
    *(v8h*)dst       = lo;
    *(v8h*)(dst + 8) = hi;
}

// ---------------------------------------------------------------------------
// K1: dec_U[b][f] = sum_d dec_last[b][d] * U_a[d][f]   (f32, tiny GEMM)
// 128 blocks: b = blk>>2, f = (blk&3)*256 + tid. U reads coalesced over f.
// ---------------------------------------------------------------------------
__global__ __launch_bounds__(256)
void k1_decU(const float* __restrict__ dec, const float* __restrict__ U,
             float* __restrict__ decU) {
    int b = blockIdx.x >> 2;
    int f = (blockIdx.x & 3) * 256 + threadIdx.x;
    const float* dl = dec + ((size_t)b * kTD + (kTD - 1)) * kD;  // last decoder step
    float acc = 0.f;
#pragma unroll 8
    for (int d = 0; d < kD; ++d)
        acc = fmaf(dl[d], U[(size_t)d * kD + f], acc);
    decU[b * kD + f] = acc;
}

// ---------------------------------------------------------------------------
// K2: fused scores[b][t] = sum_f tanh( (enc[b] @ W)[t][f] + decU[b][f] ) * Va[f]
// One block per (b, 64-row tile): 1024 blocks, 8 waves of 32.
// 64 enc rows staged to LDS as f16 **in A-fragment order** (lo/hi halves in
// separate 64KB regions -> 16B-aligned ds_load_b128). Each B fragment (from
// the pre-tiled f16 W) feeds 4 WMMAs (4 M-tiles): ~64 FLOP per L2 byte,
// B L2 traffic ~2.1 GB total. Per-wave row partials go to a workspace array
// (no atomics); K3 sums the 8 wave partials.
// ---------------------------------------------------------------------------
__global__ __launch_bounds__(256)
void k2_scores(const float* __restrict__ enc, const _Float16* __restrict__ Wt,
               const float* __restrict__ decU, const float* __restrict__ Va,
               float* __restrict__ partial) {
    // [region 0 = K-halves 0..7/8..15 "lo", region 1 = "hi"]
    // fragment index = mtile*32 + kstep; per fragment: 32 lanes x 8 halves
    __shared__ _Float16 ldsA[2][kMtiles * kKsteps * 32 * 8];  // 2 x 64 KB = 128 KB

    const int tid = threadIdx.x;
    const int b   = blockIdx.x >> 5;          // 0..31
    const int t0  = (blockIdx.x & 31) << 6;   // row tile base in TE (64 rows)

    // ---- Stage A: 64 rows x 1024 f32 -> f16 fragments in LDS ----
    // A-fragment layout (16-bit A 16x32):
    //   lane = m + 16*((kk>>3)&1); lo half h=kk&7 for kk<16, hi half for kk>=16
    const float4* src = (const float4*)(enc + ((size_t)b * kTE + t0) * kD);
    for (int q = tid; q < kMrows * kD / 4; q += 256) {    // 64 iters/thread
        float4 v   = src[q];
        int row    = q >> 8;              // 0..63
        int col    = (q & 255) << 2;      // 0..1020, multiple of 4
        int mtile  = row >> 4, m = row & 15;
        int ks     = col >> 5, kk = col & 31;
        int lane   = m | (((kk >> 3) & 1) << 4);
        int region = (kk >> 4) & 1;
        int pos    = kk & 7;              // 0 or 4; 4 elems stay in one 8-group
        _Float16* dst = &ldsA[region][(((size_t)mtile * kKsteps + ks) * 32 + lane) * 8 + pos];
        dst[0]=(_Float16)v.x; dst[1]=(_Float16)v.y;
        dst[2]=(_Float16)v.z; dst[3]=(_Float16)v.w;
    }
    __syncthreads();

    const int lane = tid & 31;
    const int wave = tid >> 5;

    float sp[kMtiles][8];
#pragma unroll
    for (int mt = 0; mt < kMtiles; ++mt)
#pragma unroll
        for (int r = 0; r < 8; ++r) sp[mt][r] = 0.f;

    for (int nt = 0; nt < 8; ++nt) {               // 8 N-tiles per wave
        const int ntile = wave * 8 + nt;
        v8f c[kMtiles];
#pragma unroll
        for (int mt = 0; mt < kMtiles; ++mt) c[mt] = (v8f){};
#pragma unroll 2
        for (int ks = 0; ks < kKsteps; ++ks) {
            // B fragment: lane L holds column N=L%16, K in table order over elements
            const _Float16* bp = Wt +
                ((((size_t)ks * kNtiles + ntile) * 32) << 4);
            v16h bm;
#pragma unroll
            for (int s = 0; s < 16; ++s)
                bm[s] = bp[((((s >> 3) << 4) + ((lane >> 4) << 3) + (s & 7)) << 4) + (lane & 15)];
            if (ks < kKsteps - 1)                    // gfx1250 global_prefetch_b8
                __builtin_prefetch(bp + ((size_t)kNtiles * 32 << 4), 0, 3);
            // A fragments for all 4 M-tiles: 16B lo + 16B hi per lane
#pragma unroll
            for (int mt = 0; mt < kMtiles; ++mt) {
                const size_t fi = (((size_t)mt * kKsteps + ks) * 32 + lane) * 8;
                v8h alo = *(const v8h*)&ldsA[0][fi];
                v8h ahi = *(const v8h*)&ldsA[1][fi];
                v16h a  = __builtin_shufflevector(alo, ahi,
                            0,1,2,3,4,5,6,7,8,9,10,11,12,13,14,15);
                c[mt] = __builtin_amdgcn_wmma_f32_16x16x32_f16(
                            false, a, false, bm, (short)0, c[mt], false, false);
            }
        }
        // Epilogue in C layout: VGPR r = row r (lanes 0-15) / row r+8 (16-31),
        // lane%16 = column within tile. decU/Va depend only on the column.
        const int col = (ntile << 4) + (lane & 15);
        const float du = decU[b * kD + col];
        const float va = Va[col];
#pragma unroll
        for (int mt = 0; mt < kMtiles; ++mt)
#pragma unroll
            for (int r = 0; r < 8; ++r)
                sp[mt][r] = fmaf(tanhf(c[mt][r] + du), va, sp[mt][r]);
    }

    // Reduce over the 16 columns held in each half-wave
#pragma unroll
    for (int mt = 0; mt < kMtiles; ++mt)
#pragma unroll
        for (int r = 0; r < 8; ++r) {
            float p = sp[mt][r];
            p += __shfl_xor(p, 1);
            p += __shfl_xor(p, 2);
            p += __shfl_xor(p, 4);
            p += __shfl_xor(p, 8);
            sp[mt][r] = p;
        }
    // lane 0 holds rows r, lane 16 holds rows r+8 (per M-tile). Write per-wave
    // partials; K3 sums the 8 waves. No atomics, no zero-init required.
    if ((lane & 15) == 0) {
        const int off = (lane >> 4) << 3;   // 0 or 8
        float* dst = partial + ((size_t)blockIdx.x * 8 + wave) * kMrows;
#pragma unroll
        for (int mt = 0; mt < kMtiles; ++mt)
#pragma unroll
            for (int r = 0; r < 8; ++r)
                dst[mt * 16 + off + r] = sp[mt][r];   // row t0 + mt*16 + off + r
    }
}

// ---------------------------------------------------------------------------
// K3: per-batch softmax over TE + context = enc^T @ weights.
// One block per b (32 blocks, 256 threads). Sums the 8 per-wave score
// partials while loading. enc reads coalesced float4.
// ---------------------------------------------------------------------------
__global__ __launch_bounds__(256)
void k3_softmax_ctx(const float* __restrict__ enc, const float* __restrict__ partial,
                    float* __restrict__ out) {
    __shared__ float red[256];
    __shared__ float wbuf[kTE];
    const int b = blockIdx.x, tid = threadIdx.x;

    float s[8];
    float m = -3.0e38f;
#pragma unroll
    for (int i = 0; i < 8; ++i) {
        const int t  = tid + i * 256;
        const int wg = b * (kTE / kMrows) + (t >> 6);   // source workgroup in K2
        const int rw = t & 63;                          // row within that WG
        const float* p = partial + ((size_t)wg * 8) * kMrows + rw;
        float acc = 0.f;
#pragma unroll
        for (int w = 0; w < 8; ++w) acc += p[w * kMrows];
        s[i] = acc;
        m = fmaxf(m, acc);
    }
    red[tid] = m; __syncthreads();
    for (int o = 128; o > 0; o >>= 1) {
        if (tid < o) red[tid] = fmaxf(red[tid], red[tid + o]);
        __syncthreads();
    }
    m = red[0]; __syncthreads();

    float e[8], sum = 0.f;
#pragma unroll
    for (int i = 0; i < 8; ++i) { e[i] = __expf(s[i] - m); sum += e[i]; }
    red[tid] = sum; __syncthreads();
    for (int o = 128; o > 0; o >>= 1) {
        if (tid < o) red[tid] += red[tid + o];
        __syncthreads();
    }
    const float inv = 1.0f / red[0];

#pragma unroll
    for (int i = 0; i < 8; ++i) {
        const int t = tid + i * 256;
        const float w = e[i] * inv;
        wbuf[t] = w;
        out[kB * kD + (size_t)b * kTE + t] = w;   // weights output region
    }
    __syncthreads();

    // context[b][col] = sum_t w[t] * enc[b][t][col], 4 cols per thread
    float4 acc = make_float4(0.f, 0.f, 0.f, 0.f);
    const float4* ep = (const float4*)(enc + (size_t)b * kTE * kD) + tid;
#pragma unroll 4
    for (int t = 0; t < kTE; ++t) {
        const float w = wbuf[t];
        float4 ev = ep[(size_t)t * (kD / 4)];
        acc.x = fmaf(w, ev.x, acc.x);
        acc.y = fmaf(w, ev.y, acc.y);
        acc.z = fmaf(w, ev.z, acc.z);
        acc.w = fmaf(w, ev.w, acc.w);
    }
    *(float4*)(out + (size_t)b * kD + tid * 4) = acc;
}

// ---------------------------------------------------------------------------
// Launch
// ---------------------------------------------------------------------------
extern "C" void kernel_launch(void* const* d_in, const int* in_sizes, int n_in,
                              void* d_out, int out_size, void* d_ws, size_t ws_size,
                              hipStream_t stream) {
    const float* enc = (const float*)d_in[0];   // (32, 2048, 1024)
    const float* dec = (const float*)d_in[1];   // (32, 64, 1024)
    const float* Wa  = (const float*)d_in[2];   // (1024, 1024)
    const float* Ua  = (const float*)d_in[3];   // (1024, 1024)
    const float* Va  = (const float*)d_in[4];   // (1024, 1)
    float* out = (float*)d_out;                 // context (32*1024) ++ weights (32*2048)

    char* ws = (char*)d_ws;
    _Float16* Wt   = (_Float16*)ws;                              // 2 MB f16 tiled W
    float* decU    = (float*)(ws + (size_t)2 * 1024 * 1024);     // 128 KB
    float* partial = (float*)(ws + (size_t)2 * 1024 * 1024 + 131072); // 2 MB partials

    k0_tile_w<<<(kD * kNtiles) / 256, 256, 0, stream>>>(Wa, Wt);
    k1_decU<<<kB * 4, 256, 0, stream>>>(dec, Ua, decU);
    k2_scores<<<kWGs, 256, 0, stream>>>(enc, Wt, decU, Va, partial);
    k3_softmax_ctx<<<kB, 256, 0, stream>>>(enc, partial, out);
}
